// SpectralConv2d_58368605552851
// MI455X (gfx1250) — matrix operand order repaired and checked by reading, not compiled, not run on hardware
//
#include <hip/hip_runtime.h>
#include <hip/hip_bf16.h>

typedef __attribute__((ext_vector_type(16))) _Float16 v16h;
typedef __attribute__((ext_vector_type(8)))  float    v8f;

#define PI2F 6.283185307179586476925f

__device__ __forceinline__ v8f wmma_f16(v16h a, v16h b, v8f c) {
  return __builtin_amdgcn_wmma_f32_16x16x32_f16(false, a, false, b, (short)0, c, false, false);
}

// 16x32 (MxK) f16 A-tile from LDS (row-major, leading dim ld) -> WMMA A layout.
// lanes 0-15: row=lane, K = 0..15 ; lanes 16-31: row=lane-16, K = 16..31
// Per lane: one contiguous 16-half (32 B) run -> ds_load_b128 x2.
__device__ __forceinline__ v16h load_A(const _Float16* s, int ld, int lane) {
  int row = lane & 15;
  int kb  = (lane >> 4) << 4;
  const _Float16* p = s + row * ld + kb;
  v16h a;
#pragma unroll
  for (int j = 0; j < 16; ++j) a[j] = p[j];
  return a;
}

// 32x16 (KxN) f16 B-tile from LDS stored TRANSPOSED ([N][K], leading dim ldk).
// s points at &sBt[n0*ldk + k0]. lanes 0-15: col=lane, K=0..15 ; lanes 16-31: col=lane-16, K=16..31.
// Per lane: one contiguous 16-half (32 B) run -> ds_load_b128 x2.
__device__ __forceinline__ v16h load_Bt(const _Float16* s, int ldk, int lane) {
  int col = lane & 15;
  int kb  = (lane >> 4) << 4;
  const _Float16* p = s + col * ldk + kb;
  v16h b;
#pragma unroll
  for (int j = 0; j < 16; ++j) b[j] = p[j];
  return b;
}

// ---------------------------------------------------------------------------
// Init: DFT matrices (f16, ortho norm). F = forward [2m][N] rows interleaved
// (re,im) = (cos, -sin)/sqrt(N). G = inverse [N][32] with c_k = (k?2:1),
// (re,im) cols = (c*cos, -c*sin)/sqrt(N); Gz zero-padded to K=32.
// ---------------------------------------------------------------------------
__global__ void init_dft(_Float16* F64, _Float16* G64, _Float16* Fz, _Float16* Gz) {
  int tid = threadIdx.x;
  const float s64 = 0.125f;                // 1/sqrt(64)
  const float s32 = 0.17677669529663688f;  // 1/sqrt(32)
  for (int t = tid; t < 32 * 64; t += 256) {      // F64 [32][64]
    int q = t >> 6, z = t & 63, k = q >> 1, ri = q & 1;
    float ang = PI2F * (float)(k * z) / 64.0f;
    F64[t] = (_Float16)(((ri == 0) ? __cosf(ang) : -__sinf(ang)) * s64);
  }
  for (int t = tid; t < 64 * 32; t += 256) {      // G64 [64][32]
    int z = t >> 5, q = t & 31, k = q >> 1, ri = q & 1;
    float ck = (k == 0) ? 1.0f : 2.0f;
    float ang = PI2F * (float)(k * z) / 64.0f;
    G64[t] = (_Float16)(((ri == 0) ? ck * __cosf(ang) : -ck * __sinf(ang)) * s64);
  }
  for (int t = tid; t < 16 * 32; t += 256) {      // Fz [16][32]
    int q = t >> 5, z = t & 31, k = q >> 1, ri = q & 1;
    float ang = PI2F * (float)(k * z) / 32.0f;
    Fz[t] = (_Float16)(((ri == 0) ? __cosf(ang) : -__sinf(ang)) * s32);
  }
  for (int t = tid; t < 32 * 32; t += 256) {      // Gz [32][32] (cols 16..31 = 0)
    int z = t >> 5, q = t & 31;
    float v = 0.0f;
    if (q < 16) {
      int k = q >> 1, ri = q & 1;
      float ck = (k == 0) ? 1.0f : 2.0f;
      float ang = PI2F * (float)(k * z) / 32.0f;
      v = ((ri == 0) ? ck * __cosf(ang) : -ck * __sinf(ang)) * s32;
    }
    Gz[t] = (_Float16)v;
  }
}

// Repack complex spectral weights fw[I=64][O=64][m][2] into per-mode f16
// concat matrices stored TRANSPOSED for the B operand:
//   dst[mode][0][o=64][ii=128] = ([Wr ; -Wi])^T , dst[mode][1][o][ii] = ([Wi ; Wr])^T
__global__ void init_wcat(const float* __restrict__ fw, int m, _Float16* __restrict__ dst) {
  int t = blockIdx.x * 256 + threadIdx.x;
  if (t >= m * 8192) return;
  int mode = t >> 13;
  int rem  = t & 8191;
  int ii = rem >> 6, o = rem & 63, i = ii & 63;
  const float* p = fw + (((size_t)i * 64 + o) * m + mode) * 2;
  float re = p[0], im = p[1];
  _Float16* dm = dst + (size_t)mode * 16384;
  dm[o * 128 + ii]        = (_Float16)((ii < 64) ? re : -im);   // real-part matrix
  dm[8192 + o * 128 + ii] = (_Float16)((ii < 64) ? im : re);    // imag-part matrix
}

// f16 conversions of conv_w [o][i], w1 [f][o], w2 [o][f] (original layouts ARE
// the transposed-B layouts we want in LDS).
__global__ void init_misc(const float* __restrict__ conv_w, const float* __restrict__ w1,
                          const float* __restrict__ w2, _Float16* cwh,
                          _Float16* w1h, _Float16* w2h) {
  int t = blockIdx.x * 256 + threadIdx.x;
  if (t < 4096)       cwh[t] = (_Float16)conv_w[t];
  else if (t < 12288) w1h[t - 4096] = (_Float16)w1[t - 4096];
  else if (t < 20480) w2h[t - 12288] = (_Float16)w2[t - 12288];
}

// ---------------------------------------------------------------------------
// acc[p,o] = sum_i x[p,i]*conv_w[o,i] + conv_b[o]   (seeds the accumulator)
// ---------------------------------------------------------------------------
__global__ void conv_mix(const float* __restrict__ x, const _Float16* __restrict__ cwh,
                         const float* __restrict__ conv_b, float* __restrict__ acc) {
  __shared__ _Float16 sA[64 * 72];   // [p][i], ld 72
  __shared__ _Float16 sBt[64 * 72];  // [o][i], ld 72  (transposed B)
  int tid = threadIdx.x;
  size_t p0 = (size_t)blockIdx.x * 64;
  for (int t = tid; t < 4096; t += 256) {
    int r = t >> 6, c = t & 63;
    sA[r * 72 + c]  = (_Float16)x[p0 * 64 + t];
    sBt[r * 72 + c] = cwh[t];
  }
  __syncthreads();
  int wave = tid >> 5, lane = tid & 31;
#pragma unroll
  for (int j = 0; j < 2; ++j) {
    int ct = wave * 2 + j;
    int tm = ct >> 2, tn = ct & 3;
    v8f c = {};
    c = wmma_f16(load_A(sA + tm * 16 * 72,      72, lane), load_Bt(sBt + tn * 16 * 72,      72, lane), c);
    c = wmma_f16(load_A(sA + tm * 16 * 72 + 32, 72, lane), load_Bt(sBt + tn * 16 * 72 + 32, 72, lane), c);
    int col  = tn * 16 + (lane & 15);
    int row0 = tm * 16 + ((lane >> 4) << 3);
    float bias = conv_b[col];
#pragma unroll
    for (int v = 0; v < 8; ++v) acc[(p0 + row0 + v) * 64 + col] = c[v] + bias;
  }
}

// ---------------------------------------------------------------------------
// Forward DFT along one axis: per line-block, Xf[bl][q][i] = F[q][:] . src_block[:, i]
// src block rows strided by rowStride; block base from bl = (i2*K1 + i1)*K0 + i0.
// ---------------------------------------------------------------------------
__global__ void dft_fwd(const float* __restrict__ src, const _Float16* __restrict__ F,
                        _Float16* __restrict__ Xf, int N, int m2, int rowStride,
                        int K0, int M0, int K1, long M1, long M2) {
  __shared__ _Float16 sF[32 * 72];    // A: [q][axis], ld N+8
  __shared__ _Float16 sInT[64 * 72];  // B transposed: [ch][axis], ld N+8
  int bl = blockIdx.x, tid = threadIdx.x;
  int ldk = N + 8;
  int i0 = bl % K0; int r = bl / K0; int i1 = r % K1; long i2 = r / K1;
  long base = (long)i0 * M0 + (long)i1 * M1 + i2 * M2;
  for (int t = tid; t < N * 64; t += 256) {
    int rr = t >> 6, cc = t & 63;                         // rr = axis pos, cc = channel
    sInT[cc * ldk + rr] = (_Float16)src[base + (long)rr * rowStride + cc];
  }
  for (int t = tid; t < m2 * N; t += 256) {
    int rr = t / N, cc = t - rr * N;
    sF[rr * ldk + cc] = F[t];
  }
  __syncthreads();
  int wave = tid >> 5, lane = tid & 31;
  int ntiles = (m2 >> 4) * 4;
  for (int ct = wave; ct < ntiles; ct += 8) {
    int tm = ct >> 2, tn = ct & 3;
    v8f c = {};
    for (int k0 = 0; k0 < N; k0 += 32)
      c = wmma_f16(load_A(sF + tm * 16 * ldk + k0, ldk, lane),
                   load_Bt(sInT + tn * 16 * ldk + k0, ldk, lane), c);
    int col  = tn * 16 + (lane & 15);
    int row0 = tm * 16 + ((lane >> 4) << 3);
    long ob = (long)bl * m2 * 64;
#pragma unroll
    for (int v = 0; v < 8; ++v) Xf[ob + (long)(row0 + v) * 64 + col] = (_Float16)c[v];
  }
}

// ---------------------------------------------------------------------------
// Per-mode complex channel mix: [Or|Oi][bl,o] = [Xr|Xi][bl,:128] . Wcat_{r,i}
// ---------------------------------------------------------------------------
__global__ void chan_mix(const _Float16* __restrict__ Xf, const _Float16* __restrict__ Wcat,
                         _Float16* __restrict__ O, int m2) {
  __shared__ _Float16 sA[64 * 136];   // [bl][128], ld 136
  __shared__ _Float16 sBr[64 * 136];  // transposed: [o][128], ld 136
  __shared__ _Float16 sBi[64 * 136];
  int tid = threadIdx.x;
  int mode = blockIdx.y;
  long t0 = (long)blockIdx.x * 64;
  int bs = m2 * 64;
  for (int t = tid; t < 64 * 128; t += 256) {
    int rr = t >> 7, cc = t & 127;
    sA[rr * 136 + cc] = Xf[(t0 + rr) * bs + mode * 128 + cc];
  }
  const _Float16* wsrc = Wcat + (long)mode * 16384;   // pre-transposed [o][128]
  for (int t = tid; t < 8192; t += 256) {
    int rr = t >> 7, cc = t & 127;
    sBr[rr * 136 + cc] = wsrc[t];
    sBi[rr * 136 + cc] = wsrc[8192 + t];
  }
  __syncthreads();
  int wave = tid >> 5, lane = tid & 31;
  int ri = wave & 1, tm4 = wave >> 1;
  const _Float16* sB = ri ? sBi : sBr;
  v16h a0 = load_A(sA + tm4 * 16 * 136 + 0,  136, lane);
  v16h a1 = load_A(sA + tm4 * 16 * 136 + 32, 136, lane);
  v16h a2 = load_A(sA + tm4 * 16 * 136 + 64, 136, lane);
  v16h a3 = load_A(sA + tm4 * 16 * 136 + 96, 136, lane);
#pragma unroll
  for (int tn = 0; tn < 4; ++tn) {
    const _Float16* bt = sB + tn * 16 * 136;
    v8f c = {};
    c = wmma_f16(a0, load_Bt(bt + 0,  136, lane), c);
    c = wmma_f16(a1, load_Bt(bt + 32, 136, lane), c);
    c = wmma_f16(a2, load_Bt(bt + 64, 136, lane), c);
    c = wmma_f16(a3, load_Bt(bt + 96, 136, lane), c);
    int col  = tn * 16 + (lane & 15);
    int row0 = tm4 * 16 + ((lane >> 4) << 3);
#pragma unroll
    for (int v = 0; v < 8; ++v)
      O[(t0 + row0 + v) * bs + mode * 128 + ri * 64 + col] = (_Float16)c[v];
  }
}

// ---------------------------------------------------------------------------
// Inverse DFT + accumulate: acc_block[z,o] += G[z][:32] . O_block[:32,o]
// (K zero-padded to 32 when m2 == 16). Strided f32 RMW on acc.
// ---------------------------------------------------------------------------
__global__ void dft_inv(const _Float16* __restrict__ O, const _Float16* __restrict__ G,
                        float* __restrict__ acc, int N, int m2, int rowStride,
                        int K0, int M0, int K1, long M1, long M2) {
  __shared__ _Float16 sG[64 * 40];   // A: [z][q<=32], ld 40
  __shared__ _Float16 sOT[64 * 40];  // B transposed: [o][q], ld 40
  int bl = blockIdx.x, tid = threadIdx.x;
  int i0 = bl % K0; int r = bl / K0; int i1 = r % K1; long i2 = r / K1;
  long base = (long)i0 * M0 + (long)i1 * M1 + i2 * M2;
  for (int t = tid; t < N * 32; t += 256) {
    int rr = t >> 5, cc = t & 31;
    sG[rr * 40 + cc] = G[t];
  }
  for (int t = tid; t < 32 * 64; t += 256) {
    int q = t >> 6, o = t & 63;
    sOT[o * 40 + q] = (q < m2) ? O[(long)bl * m2 * 64 + t] : (_Float16)0.0f;
  }
  __syncthreads();
  int wave = tid >> 5, lane = tid & 31;
  int ntiles = (N >> 4) * 4;
  for (int ct = wave; ct < ntiles; ct += 8) {
    int tm = ct >> 2, tn = ct & 3;
    int col  = tn * 16 + (lane & 15);
    int row0 = tm * 16 + ((lane >> 4) << 3);
    v8f c;
#pragma unroll
    for (int v = 0; v < 8; ++v) c[v] = acc[base + (long)(row0 + v) * rowStride + col];
    c = wmma_f16(load_A(sG + tm * 16 * 40, 40, lane), load_Bt(sOT + tn * 16 * 40, 40, lane), c);
#pragma unroll
    for (int v = 0; v < 8; ++v) acc[base + (long)(row0 + v) * rowStride + col] = c[v];
  }
}

// ---------------------------------------------------------------------------
// Fused FFN: out = relu(xp @ w1^T + b1) @ w2^T + b2 ; hidden tile lives in LDS.
// ---------------------------------------------------------------------------
__global__ void ffn(const float* __restrict__ acc, const _Float16* __restrict__ w1h,
                    const float* __restrict__ b1, const _Float16* __restrict__ w2h,
                    const float* __restrict__ b2, float* __restrict__ out) {
  __shared__ _Float16 sX[64 * 72];     // [p][o], ld 72
  __shared__ _Float16 sW1t[128 * 72];  // transposed B1: [f][o], ld 72 (= w1 layout)
  __shared__ _Float16 sH[64 * 136];    // [p][f], ld 136 (A of GEMM2)
  __shared__ _Float16 sW2t[64 * 136];  // transposed B2: [o][f], ld 136 (= w2 layout)
  int tid = threadIdx.x;
  size_t p0 = (size_t)blockIdx.x * 64;
  for (int t = tid; t < 4096; t += 256) {
    int r = t >> 6, c = t & 63;
    sX[r * 72 + c] = (_Float16)acc[p0 * 64 + t];
  }
  for (int t = tid; t < 8192; t += 256) {
    int r1 = t >> 6, c1 = t & 63;
    sW1t[r1 * 72 + c1] = w1h[t];
    int r2 = t >> 7, c2 = t & 127;
    sW2t[r2 * 136 + c2] = w2h[t];
  }
  __syncthreads();
  int wave = tid >> 5, lane = tid & 31;
  // GEMM1: [64 x 64] x [64 x 128] -> relu -> sH (f16)
#pragma unroll
  for (int j = 0; j < 4; ++j) {
    int ct = wave * 4 + j;
    int tm = ct >> 3, tn = ct & 7;
    v8f c = {};
    c = wmma_f16(load_A(sX + tm * 16 * 72,      72, lane), load_Bt(sW1t + tn * 16 * 72,      72, lane), c);
    c = wmma_f16(load_A(sX + tm * 16 * 72 + 32, 72, lane), load_Bt(sW1t + tn * 16 * 72 + 32, 72, lane), c);
    int col  = tn * 16 + (lane & 15);
    int row0 = tm * 16 + ((lane >> 4) << 3);
    float bias = b1[col];
#pragma unroll
    for (int v = 0; v < 8; ++v) {
      float h = c[v] + bias;
      sH[(row0 + v) * 136 + col] = (_Float16)(h > 0.0f ? h : 0.0f);
    }
  }
  __syncthreads();
  // GEMM2: [64 x 128] x [128 x 64] -> out
#pragma unroll
  for (int j = 0; j < 2; ++j) {
    int ct = wave * 2 + j;
    int tm = ct >> 2, tn = ct & 3;
    v8f c = {};
#pragma unroll
    for (int k0 = 0; k0 < 128; k0 += 32)
      c = wmma_f16(load_A(sH + tm * 16 * 136 + k0, 136, lane),
                   load_Bt(sW2t + tn * 16 * 136 + k0, 136, lane), c);
    int col  = tn * 16 + (lane & 15);
    int row0 = tm * 16 + ((lane >> 4) << 3);
    float bias = b2[col];
#pragma unroll
    for (int v = 0; v < 8; ++v) out[(p0 + row0 + v) * 64 + col] = c[v] + bias;
  }
}

// ---------------------------------------------------------------------------
extern "C" void kernel_launch(void* const* d_in, const int* in_sizes, int n_in,
                              void* d_out, int out_size, void* d_ws, size_t ws_size,
                              hipStream_t stream) {
  (void)in_sizes; (void)n_in; (void)out_size; (void)ws_size;
  const float* x      = (const float*)d_in[0];
  const float* x_raw  = (const float*)d_in[1];
  const float* fw_x   = (const float*)d_in[2];
  const float* fw_y   = (const float*)d_in[3];
  const float* fw_z   = (const float*)d_in[4];
  const float* fwr_x  = (const float*)d_in[5];
  const float* fwr_y  = (const float*)d_in[6];
  const float* conv_w = (const float*)d_in[7];
  const float* conv_b = (const float*)d_in[8];
  const float* w1     = (const float*)d_in[9];
  const float* b1     = (const float*)d_in[10];
  const float* w2     = (const float*)d_in[11];
  const float* b2     = (const float*)d_in[12];
  float* out = (float*)d_out;

  // workspace layout
  char* ws = (char*)d_ws;
  constexpr size_t OFF_ACC  = 0;                        // 524288*64 f32 = 128 MiB
  constexpr size_t OFF_XF   = OFF_ACC + 134217728;      // 16 MiB
  constexpr size_t OFF_O    = OFF_XF + 16777216;        // 16 MiB
  constexpr size_t OFF_F64  = OFF_O + 16777216;
  constexpr size_t OFF_G64  = OFF_F64 + 4096;
  constexpr size_t OFF_FZ   = OFF_G64 + 4096;
  constexpr size_t OFF_GZ   = OFF_FZ + 1024;
  constexpr size_t OFF_WCAT = OFF_GZ + 2048;            // 72 modes * 32 KiB
  constexpr size_t OFF_CW   = OFF_WCAT + 2359296;
  constexpr size_t OFF_W1T  = OFF_CW + 8192;
  constexpr size_t OFF_W2T  = OFF_W1T + 16384;

  float*     acc  = (float*)(ws + OFF_ACC);
  _Float16*  Xf   = (_Float16*)(ws + OFF_XF);
  _Float16*  Obuf = (_Float16*)(ws + OFF_O);
  _Float16*  F64  = (_Float16*)(ws + OFF_F64);
  _Float16*  G64  = (_Float16*)(ws + OFF_G64);
  _Float16*  Fz   = (_Float16*)(ws + OFF_FZ);
  _Float16*  Gz   = (_Float16*)(ws + OFF_GZ);
  _Float16*  Wz   = (_Float16*)(ws + OFF_WCAT);              // 8 modes
  _Float16*  Wy   = Wz  + (size_t)8  * 16384;                // 16 modes each below
  _Float16*  Wyr  = Wy  + (size_t)16 * 16384;
  _Float16*  Wx   = Wyr + (size_t)16 * 16384;
  _Float16*  Wxr  = Wx  + (size_t)16 * 16384;
  _Float16*  cwh  = (_Float16*)(ws + OFF_CW);
  _Float16*  w1h  = (_Float16*)(ws + OFF_W1T);
  _Float16*  w2h  = (_Float16*)(ws + OFF_W2T);

  // init (recomputed every call -> deterministic)
  init_dft<<<1, 256, 0, stream>>>(F64, G64, Fz, Gz);
  init_wcat<<<(8 * 8192) / 256, 256, 0, stream>>>(fw_z, 8, Wz);
  init_wcat<<<(16 * 8192) / 256, 256, 0, stream>>>(fw_y, 16, Wy);
  init_wcat<<<(16 * 8192) / 256, 256, 0, stream>>>(fwr_y, 16, Wyr);
  init_wcat<<<(16 * 8192) / 256, 256, 0, stream>>>(fw_x, 16, Wx);
  init_wcat<<<(16 * 8192) / 256, 256, 0, stream>>>(fwr_x, 16, Wxr);
  init_misc<<<80, 256, 0, stream>>>(conv_w, w1, w2, cwh, w1h, w2h);

  // acc = 1x1 conv (channel mix) + bias
  conv_mix<<<8192, 256, 0, stream>>>(x, cwh, conv_b, acc);

  // z-axis spectral conv (x only): blocks = (b,s1,s2) contiguous; z-stride 64
  dft_fwd<<<8192, 256, 0, stream>>>(x, Fz, Xf, 32, 16, 64, 8192, 2048, 1, 0L, 0L);
  chan_mix<<<dim3(128, 8), 256, 0, stream>>>(Xf, Wz, Obuf, 16);
  dft_inv<<<8192, 256, 0, stream>>>(Obuf, Gz, acc, 32, 16, 64, 8192, 2048, 1, 0L, 0L);

  // y-axis spectral conv: blocks = (b,s1,s3); row stride (y) = 2048
  dft_fwd<<<4096, 256, 0, stream>>>(x, F64, Xf, 64, 32, 2048, 32, 64, 64, 131072L, 8388608L);
  chan_mix<<<dim3(64, 16), 256, 0, stream>>>(Xf, Wy, Obuf, 32);
  dft_inv<<<4096, 256, 0, stream>>>(Obuf, G64, acc, 64, 32, 2048, 32, 64, 64, 131072L, 8388608L);

  dft_fwd<<<4096, 256, 0, stream>>>(x_raw, F64, Xf, 64, 32, 2048, 32, 64, 64, 131072L, 8388608L);
  chan_mix<<<dim3(64, 16), 256, 0, stream>>>(Xf, Wyr, Obuf, 32);
  dft_inv<<<4096, 256, 0, stream>>>(Obuf, G64, acc, 64, 32, 2048, 32, 64, 64, 131072L, 8388608L);

  // x-axis spectral conv: blocks = (b,s2,s3); row stride (x) = 131072
  dft_fwd<<<4096, 256, 0, stream>>>(x, F64, Xf, 64, 32, 131072, 32, 64, 64, 2048L, 8388608L);
  chan_mix<<<dim3(64, 16), 256, 0, stream>>>(Xf, Wx, Obuf, 32);
  dft_inv<<<4096, 256, 0, stream>>>(Obuf, G64, acc, 64, 32, 131072, 32, 64, 64, 2048L, 8388608L);

  dft_fwd<<<4096, 256, 0, stream>>>(x_raw, F64, Xf, 64, 32, 131072, 32, 64, 64, 2048L, 8388608L);
  chan_mix<<<dim3(64, 16), 256, 0, stream>>>(Xf, Wxr, Obuf, 32);
  dft_inv<<<4096, 256, 0, stream>>>(Obuf, G64, acc, 64, 32, 131072, 32, 64, 64, 2048L, 8388608L);

  // fused FFN -> out
  ffn<<<8192, 256, 0, stream>>>(acc, w1h, b1, w2h, b2, out);
}
